// SparseMoE_44074954392151
// MI455X (gfx1250) — matrix operand order repaired and checked by reading, not compiled, not run on hardware
//
#include <hip/hip_runtime.h>
#include <hip/hip_bf16.h>

#define N_TOK 4096   // B*T
#define DIM   1024   // D
#define NEXP  8      // E
#define HID   2048   // H

typedef __attribute__((ext_vector_type(16))) __bf16 v16bf;
typedef __attribute__((ext_vector_type(8)))  float  v8f;

__device__ __forceinline__ unsigned short f2bf(float f) {
  unsigned int u = __float_as_uint(f);
  u += 0x7fffu + ((u >> 16) & 1u);   // round-to-nearest-even
  return (unsigned short)(u >> 16);
}

union FragU { v16bf v; uint4 q[2]; };

// A fragment: 16x32 bf16, row-major [16][ldA] in LDS at column k0.
// Lanes 0-15: row m=lane, K = k0+0..7 (q0) and k0+16..23 (q1)
// Lanes 16-31: row m=lane-16, K = k0+8..15 and k0+24..31
__device__ __forceinline__ v16bf load_a_lds(const unsigned short* base, int ldA, int k0, int lane) {
  const int m  = lane & 15;
  const int kb = k0 + ((lane & 16) ? 8 : 0);
  FragU f;
  f.q[0] = *(const uint4*)(base + m * ldA + kb);
  f.q[1] = *(const uint4*)(base + m * ldA + kb + 16);
  return f.v;
}

// B fragment: 32x16 bf16 (KxN). Weights stored transposed: wT[n][k], ld = K-dim.
// Lanes 0-15: col n=row0+lane, K = k0+0..7 and k0+16..23; lanes 16-31: K = k0+8..15 and k0+24..31.
__device__ __forceinline__ v16bf load_b_glb(const unsigned short* __restrict__ wT,
                                            size_t row0, int ld, int k0, int lane) {
  const int n  = lane & 15;
  const int kb = k0 + ((lane & 16) ? 8 : 0);
  FragU f;
  const unsigned short* p = wT + (row0 + n) * (size_t)ld + kb;
  f.q[0] = *(const uint4*)(p);
  f.q[1] = *(const uint4*)(p + 16);
  return f.v;
}

__global__ void zero_kernel(float* __restrict__ out, int n, int* __restrict__ counts) {
  int i = blockIdx.x * blockDim.x + threadIdx.x;
  for (int j = i; j < n; j += gridDim.x * blockDim.x) out[j] = 0.f;
  if (i < NEXP) counts[i] = 0;
}

__global__ void cvt_x_kernel(const float* __restrict__ x, unsigned short* __restrict__ xb, int n) {
  for (int i = blockIdx.x * blockDim.x + threadIdx.x; i < n; i += gridDim.x * blockDim.x)
    xb[i] = f2bf(x[i]);
}

// w: [E][R][C] f32  ->  wt: [E][C][R] bf16  (coalesced writes)
__global__ void cvt_transpose_kernel(const float* __restrict__ w, unsigned short* __restrict__ wt,
                                     int R, int C) {
  const long long total = (long long)NEXP * R * C;
  const int rc = R * C;
  for (long long i = blockIdx.x * (long long)blockDim.x + threadIdx.x; i < total;
       i += (long long)gridDim.x * blockDim.x) {
    int e = (int)(i / rc);
    int rem = (int)(i % rc);
    int c = rem / R;
    int r = rem % R;
    wt[i] = f2bf(w[(size_t)e * rc + (size_t)r * C + c]);
  }
}

// One thread per token: logits, top-2, softmax gates, append to expert lists.
__global__ void router_kernel(const float* __restrict__ x, const float* __restrict__ Wr,
                              const float* __restrict__ br, float* __restrict__ gates,
                              int* __restrict__ counts, int* __restrict__ lists) {
  const int t = blockIdx.x * blockDim.x + threadIdx.x;
  if (t >= N_TOK) return;
  float lg[NEXP];
#pragma unroll
  for (int e = 0; e < NEXP; ++e) lg[e] = br[e];
  const float* xr = x + (size_t)t * DIM;
  for (int d = 0; d < DIM; ++d) {
    const float xv = xr[d];
#pragma unroll
    for (int e = 0; e < NEXP; ++e) lg[e] += xv * Wr[d * NEXP + e];
  }
  int i1 = 0;
#pragma unroll
  for (int e = 1; e < NEXP; ++e) if (lg[e] > lg[i1]) i1 = e;
  int i2 = (i1 == 0) ? 1 : 0;
#pragma unroll
  for (int e = 0; e < NEXP; ++e) if (e != i1 && lg[e] > lg[i2]) i2 = e;
  const float m  = lg[i1];
  const float e2 = __expf(lg[i2] - m);
  const float s  = 1.f + e2;
  const float g1 = 1.f / s, g2 = e2 / s;
#pragma unroll
  for (int e = 0; e < NEXP; ++e) gates[(size_t)t * NEXP + e] = 0.f;
  gates[(size_t)t * NEXP + i1] = g1;
  gates[(size_t)t * NEXP + i2] = g2;
  int p1 = atomicAdd(&counts[i1], 1); lists[i1 * N_TOK + p1] = t;
  int p2 = atomicAdd(&counts[i2], 1); lists[i2 * N_TOK + p2] = t;
}

// Fused expert FFN: block = (expert e, 32-token tile). 256 threads = 8 waves.
// M=32 halves expert-weight re-read traffic vs M=16: every B fragment feeds
// two WMMA accumulator tiles (two independent chains -> better latency overlap).
// LDS: X tile 32x1024 bf16 (64KB) + H chunk 32x128 bf16 (8KB).
__global__ void __launch_bounds__(256) moe_ffn_kernel(
    const unsigned short* __restrict__ xb,    // [N][D] bf16
    const unsigned short* __restrict__ w1t,   // [E][H][D] bf16 (transposed)
    const unsigned short* __restrict__ w2t,   // [E][D][H] bf16 (transposed)
    const float* __restrict__ b1,             // [E][H]
    const float* __restrict__ b2,             // [E][D]
    const float* __restrict__ gates,          // [N][E]
    const int* __restrict__ counts,
    const int* __restrict__ lists,            // [E][N]
    float* __restrict__ out)                  // [N][D]
{
  __shared__ unsigned short Xs[32 * DIM];
  __shared__ unsigned short Hs[32 * 128];
  __shared__ int   toks[32];
  __shared__ float gts[32];

  const int e    = blockIdx.y;
  const int cnt  = counts[e];
  const int row0 = blockIdx.x * 32;
  if (row0 >= cnt) return;
  const int rows = min(32, cnt - row0);

  const int tid  = threadIdx.x;
  const int lane = tid & 31;
  const int w    = tid >> 5;

  if (tid < 32) {
    const int tok = (tid < rows) ? lists[e * N_TOK + row0 + tid] : 0;
    toks[tid] = tok;
    gts[tid]  = (tid < rows) ? gates[(size_t)tok * NEXP + e] : 0.f;
  }
  __syncthreads();

  // Stage X tile into LDS (b128, coalesced; zero-pad short tiles)
  for (int i = tid; i < 32 * 128; i += 256) {
    const int r  = i >> 7;
    const int cq = i & 127;
    uint4 val = make_uint4(0u, 0u, 0u, 0u);
    if (r < rows) val = *(const uint4*)(xb + (size_t)toks[r] * DIM + cq * 8);
    *(uint4*)(Xs + r * DIM + cq * 8) = val;
  }
  __syncthreads();

  const v8f zf = {0.f, 0.f, 0.f, 0.f, 0.f, 0.f, 0.f, 0.f};
  v8f acc0[8], acc1[8];
#pragma unroll
  for (int j = 0; j < 8; ++j) { acc0[j] = zf; acc1[j] = zf; }

  const int nC = lane & 15;
  const int mC = (lane & 16) ? 8 : 0;

  for (int hc = 0; hc < HID / 128; ++hc) {
    // phase 1: this wave computes H columns [hc*128 + w*16, +16) for both
    // 16-row tiles over K=1024; one B fragment feeds two WMMAs.
    const int hn0 = hc * 128 + w * 16;
    const size_t w1row = (size_t)e * HID + hn0;
    v8f hacc0 = zf, hacc1 = zf;
#pragma unroll 4
    for (int kk = 0; kk < DIM / 32; ++kk) {
      v16bf b  = load_b_glb(w1t, w1row, DIM, kk * 32, lane);
      v16bf a0 = load_a_lds(Xs, DIM, kk * 32, lane);
      v16bf a1 = load_a_lds(Xs + 16 * DIM, DIM, kk * 32, lane);
      hacc0 = __builtin_amdgcn_wmma_f32_16x16x32_bf16(false, a0, false, b, (short)0, hacc0, false, false);
      hacc1 = __builtin_amdgcn_wmma_f32_16x16x32_bf16(false, a1, false, b, (short)0, hacc1, false, false);
    }
    const float b1v = b1[(size_t)e * HID + hn0 + nC];
#pragma unroll
    for (int r = 0; r < 8; ++r) {
      const float v0 = fmaxf(hacc0[r] + b1v, 0.f);
      const float v1 = fmaxf(hacc1[r] + b1v, 0.f);
      Hs[(mC + r) * 128 + w * 16 + nC]        = f2bf(v0);
      Hs[(16 + mC + r) * 128 + w * 16 + nC]   = f2bf(v1);
    }
    __syncthreads();

    // phase 2: acc += Htile(32x128) @ W2chunk ; this wave owns out cols [w*128, +128)
    v16bf aH0[4], aH1[4];
#pragma unroll
    for (int kk = 0; kk < 4; ++kk) {
      aH0[kk] = load_a_lds(Hs, 128, kk * 32, lane);
      aH1[kk] = load_a_lds(Hs + 16 * 128, 128, kk * 32, lane);
    }
    const size_t w2base = (size_t)e * DIM;
#pragma unroll
    for (int j = 0; j < 8; ++j) {
      const size_t w2row = w2base + w * 128 + j * 16;
#pragma unroll
      for (int kk = 0; kk < 4; ++kk) {
        v16bf b = load_b_glb(w2t, w2row, HID, hc * 128 + kk * 32, lane);
        acc0[j] = __builtin_amdgcn_wmma_f32_16x16x32_bf16(false, aH0[kk], false, b, (short)0, acc0[j], false, false);
        acc1[j] = __builtin_amdgcn_wmma_f32_16x16x32_bf16(false, aH1[kk], false, b, (short)0, acc1[j], false, false);
      }
    }
    __syncthreads();
  }

  // epilogue: out[tok][n] += gate * (acc + b2[n])
#pragma unroll
  for (int j = 0; j < 8; ++j) {
    const int n = w * 128 + j * 16 + nC;
    const float b2v = b2[(size_t)e * DIM + n];
#pragma unroll
    for (int r = 0; r < 8; ++r) {
      const int m0 = mC + r;
      const int m1 = 16 + mC + r;
      if (m0 < rows) {
        const float val = gts[m0] * (acc0[j][r] + b2v);
        atomicAdd(out + (size_t)toks[m0] * DIM + n, val);
      }
      if (m1 < rows) {
        const float val = gts[m1] * (acc1[j][r] + b2v);
        atomicAdd(out + (size_t)toks[m1] * DIM + n, val);
      }
    }
  }
}

extern "C" void kernel_launch(void* const* d_in, const int* in_sizes, int n_in,
                              void* d_out, int out_size, void* d_ws, size_t ws_size,
                              hipStream_t stream) {
  (void)in_sizes; (void)n_in; (void)out_size; (void)ws_size;
  const float* x  = (const float*)d_in[0];
  const float* Wr = (const float*)d_in[1];
  const float* br = (const float*)d_in[2];
  const float* W1 = (const float*)d_in[3];
  const float* b1 = (const float*)d_in[4];
  const float* W2 = (const float*)d_in[5];
  const float* b2 = (const float*)d_in[6];
  float* out = (float*)d_out;

  char* ws = (char*)d_ws;
  unsigned short* xb  = (unsigned short*)ws; ws += (size_t)N_TOK * DIM * 2;
  unsigned short* w1t = (unsigned short*)ws; ws += (size_t)NEXP * DIM * HID * 2;
  unsigned short* w2t = (unsigned short*)ws; ws += (size_t)NEXP * HID * DIM * 2;
  float* gates = (float*)ws;                 ws += (size_t)N_TOK * NEXP * 4;
  int* counts  = (int*)ws;                   ws += 128;
  int* lists   = (int*)ws;

  zero_kernel<<<2048, 256, 0, stream>>>(out, N_TOK * DIM, counts);
  cvt_x_kernel<<<2048, 256, 0, stream>>>(x, xb, N_TOK * DIM);
  cvt_transpose_kernel<<<4096, 256, 0, stream>>>(W1, w1t, DIM, HID);  // -> [E][H][D]
  cvt_transpose_kernel<<<4096, 256, 0, stream>>>(W2, w2t, HID, DIM);  // -> [E][D][H]
  router_kernel<<<N_TOK / 256, 256, 0, stream>>>(x, Wr, br, gates, counts, lists);

  dim3 grid(N_TOK / 32, NEXP);
  moe_ffn_kernel<<<grid, 256, 0, stream>>>(xb, w1t, w2t, b1, b2, gates, counts, lists, out);
}
